// CGRUCell_23922967838776
// MI455X (gfx1250) — matrix operand order repaired and checked by reading, not compiled, not run on hardware
//
#include <hip/hip_runtime.h>
#include <stdint.h>

// ---------------------------------------------------------------------------
// CDNA5 (gfx1250) fused complex-GRU cell.
// B=4096, D=U=1024.  GEMMs via v_wmma_f32_16x16x32_bf16 (wave32),
// double-buffered LDS tiles with async global->LDS copies when available.
// ---------------------------------------------------------------------------

typedef __attribute__((ext_vector_type(16))) __bf16 v16bf;
typedef __attribute__((ext_vector_type(8)))  float  v8f;
typedef __attribute__((ext_vector_type(4)))  int    v4i;

union Frag {
    v16bf v;
    uint4 q[2];
};

#define BDIM  4096
#define KDIM  2048   // 2*D == 2*U
#define NDIM  2048   // 2*U
#define HALF  1024   // U
#define WCOLS 3072   // 3*U
#define NKT   (KDIM / 32)

// ---- async global->LDS support (guarded; falls back to VGPR round-trip) ----
#if defined(__HIP_DEVICE_COMPILE__) && __has_builtin(__builtin_amdgcn_global_load_async_to_lds_b128)
#define USE_ASYNC_LDS 1
#else
#define USE_ASYNC_LDS 0
#endif

typedef __attribute__((address_space(1))) v4i as1_v4i;
typedef __attribute__((address_space(3))) v4i as3_v4i;

__device__ __forceinline__ void cp16_g2l(uint16_t* lds, const uint16_t* g)
{
#if USE_ASYNC_LDS
    __builtin_amdgcn_global_load_async_to_lds_b128(
        (as1_v4i*)g, (as3_v4i*)lds, 0, 0);
#else
    *(uint4*)lds = *(const uint4*)g;
#endif
}

__device__ __forceinline__ void wait_async_lds()
{
#if USE_ASYNC_LDS
#if __has_builtin(__builtin_amdgcn_s_wait_asynccnt)
    __builtin_amdgcn_s_wait_asynccnt(0);
#else
    asm volatile("s_wait_asynccnt 0" ::: "memory");
#endif
#endif
}

// ---------------------------------------------------------------------------
// fp32 -> bf16 (round to nearest even), elementwise
// ---------------------------------------------------------------------------
__global__ __launch_bounds__(256) void f32_to_bf16_kernel(
    const float* __restrict__ src, uint16_t* __restrict__ dst, int n)
{
    int i = blockIdx.x * 256 + threadIdx.x;
    if (i < n) {
        uint32_t u = __float_as_uint(src[i]);
        u = u + 0x7FFFu + ((u >> 16) & 1u);
        dst[i] = (uint16_t)(u >> 16);
    }
}

// ---------------------------------------------------------------------------
// helpers for the structured weight [[R,-I],[I,R]]
// ---------------------------------------------------------------------------
struct BSrc { const uint16_t* p; uint32_t sgn; };

__device__ __forceinline__ BSrc resolve_src(
    const uint16_t* R, const uint16_t* I,
    bool ktop, bool nleft, int kk, int g, int nn)
{
    BSrc b;
    if (ktop) { if (nleft) { b.p = R; b.sgn = 0u; } else { b.p = I; b.sgn = 0x80008000u; } }
    else      { if (nleft) { b.p = I; b.sgn = 0u; } else { b.p = R; b.sgn = 0u; } }
    b.p += (size_t)kk * WCOLS + g * HALF + nn;
    return b;
}

// transpose-pack one 32K x 64N weight tile into LDS [n][k] (pairs along K)
__device__ __forceinline__ void fill_btile(
    uint16_t* dst, const uint16_t* p, uint32_t sgn, int kp, int nc)
{
    const uint16_t* r0 = p + (size_t)(2 * kp) * WCOLS + nc * 4;
    const uint16_t* r1 = r0 + WCOLS;
    __builtin_prefetch((const void*)(r0 + (size_t)32 * WCOLS), 0, 0);
    ushort4 a0 = *(const ushort4*)r0;
    ushort4 a1 = *(const ushort4*)r1;
    #pragma unroll
    for (int e = 0; e < 4; ++e) {
        uint32_t pk = (uint32_t)((&a0.x)[e]) | ((uint32_t)((&a1.x)[e]) << 16);
        pk ^= sgn;
        *(uint32_t*)&dst[(nc * 4 + e) * 40 + 2 * kp] = pk;
    }
}

// load one 16x32 bf16 A fragment from LDS [m][k] tile
__device__ __forceinline__ void load_afrag(
    Frag& f, const uint16_t* tile, int m, int hi)
{
    const uint16_t* pa = tile + m * 40;
    f.q[0] = *(const uint4*)(pa + hi * 8);
    f.q[1] = *(const uint4*)(pa + 16 + hi * 8);
}

// load one 32x16 bf16 B fragment from LDS [n][k] tile
__device__ __forceinline__ void load_bfrag(
    Frag& f, const uint16_t* tile, int n, int hi)
{
    const uint16_t* pb = tile + n * 40 + hi * 8;
    f.q[0] = *(const uint4*)(pb);
    f.q[1] = *(const uint4*)(pb + 16);
}

// ---------------------------------------------------------------------------
// G1: fused 5-GEMM (x_z, x_r, x_h with A=inputs ; rec_z, rec_r with A=h_tm1)
// Block tile: 128(M) x 64(N), K step 32.  8 waves, each 32x32, 5 accum sets.
// Double-buffered LDS.
// ---------------------------------------------------------------------------
__global__ __launch_bounds__(256) void cgru_gates_kernel(
    const uint16_t* __restrict__ Xbf, const uint16_t* __restrict__ Hbf,
    const uint16_t* __restrict__ Wr,  const uint16_t* __restrict__ Wi,
    const uint16_t* __restrict__ Ur,  const uint16_t* __restrict__ Ui,
    const float* __restrict__ real_bias, const float* __restrict__ imag_bias,
    const float* __restrict__ h_tm1,
    float* __restrict__ Zbuf, float* __restrict__ XHbuf,
    uint16_t* __restrict__ RHbf)
{
    __shared__ uint16_t sAin[2][128 * 40];
    __shared__ uint16_t sAh [2][128 * 40];
    __shared__ uint16_t sB  [2][5][64 * 40];

    const int t    = threadIdx.x;
    const int bn0  = blockIdx.x * 64;
    const int bm0  = blockIdx.y * 128;
    const int lane = t & 31;
    const int wave = t >> 5;
    const int wm   = wave & 3;   // M sub-tile 0..3  (x32)
    const int wn   = wave >> 2;  // N sub-tile 0..1  (x32)
    const int l16  = lane & 15;
    const int hi   = lane >> 4;

    v8f acc[5][2][2];
    #pragma unroll
    for (int s = 0; s < 5; ++s)
        #pragma unroll
        for (int i = 0; i < 2; ++i)
            #pragma unroll
            for (int j = 0; j < 2; ++j)
                #pragma unroll
                for (int e = 0; e < 8; ++e) acc[s][i][j][e] = 0.0f;

    const bool nleft = (bn0 < HALF);
    const int  nn    = nleft ? bn0 : (bn0 - HALF);
    const int  kp    = t >> 4;   // k-pair 0..15
    const int  nc    = t & 15;   // n chunk-of-4 0..15

    // ---- tile fill (A async, B transpose-pack) into buffer `buf` ----
    auto fill = [&](int buf, int kt) {
        const int  k0   = kt * 32;
        const bool ktop = (k0 < HALF);
        const int  kk   = ktop ? k0 : (k0 - HALF);
        #pragma unroll
        for (int c = t; c < 512; c += 256) {
            const int row = c >> 2, q = c & 3;
            cp16_g2l(&sAin[buf][row * 40 + q * 8],
                     &Xbf[(size_t)(bm0 + row) * KDIM + k0 + q * 8]);
            cp16_g2l(&sAh[buf][row * 40 + q * 8],
                     &Hbf[(size_t)(bm0 + row) * KDIM + k0 + q * 8]);
        }
        #pragma unroll
        for (int s = 0; s < 5; ++s) {
            const int g = (s < 3) ? s : (s - 3);
            BSrc bs = resolve_src((s < 3) ? Wr : Ur, (s < 3) ? Wi : Ui,
                                  ktop, nleft, kk, g, nn);
            fill_btile(&sB[buf][s][0], bs.p, bs.sgn, kp, nc);
        }
    };

    fill(0, 0);
    wait_async_lds();
    __syncthreads();

    for (int kt = 0; kt < NKT; ++kt) {
        const int cur = kt & 1;
        if (kt + 1 < NKT) fill((kt + 1) & 1, kt + 1);

        // ---- fragments + WMMA on current buffer ----
        Frag aIn[2], aH[2];
        #pragma unroll
        for (int i = 0; i < 2; ++i) {
            const int m = wm * 32 + i * 16 + l16;
            load_afrag(aIn[i], &sAin[cur][0], m, hi);
            load_afrag(aH[i],  &sAh[cur][0],  m, hi);
        }
        #pragma unroll
        for (int s = 0; s < 5; ++s) {
            Frag bfr[2];
            #pragma unroll
            for (int j = 0; j < 2; ++j)
                load_bfrag(bfr[j], &sB[cur][s][0], wn * 32 + j * 16 + l16, hi);
            #pragma unroll
            for (int i = 0; i < 2; ++i) {
                const v16bf av = (s < 3) ? aIn[i].v : aH[i].v;
                #pragma unroll
                for (int j = 0; j < 2; ++j)
                    acc[s][i][j] = __builtin_amdgcn_wmma_f32_16x16x32_bf16(
                        false, av, false, bfr[j].v, (short)0, acc[s][i][j],
                        false, false);
            }
        }
        wait_async_lds();
        __syncthreads();
    }

    // ---- epilogue: z, x_h+b_h, r*h_tm1 ----
    #pragma unroll
    for (int j = 0; j < 2; ++j) {
        const int col = bn0 + wn * 32 + j * 16 + l16;
        const int nb  = (col < HALF) ? col : (col - HALF);
        const float* bsrc = (col < HALF) ? real_bias : imag_bias;
        const float bz = bsrc[0 * HALF + nb];
        const float br = bsrc[1 * HALF + nb];
        const float bh = bsrc[2 * HALF + nb];
        #pragma unroll
        for (int i = 0; i < 2; ++i) {
            const int mrow = bm0 + wm * 32 + i * 16 + hi * 8;
            #pragma unroll
            for (int r = 0; r < 8; ++r) {
                const size_t idx = (size_t)(mrow + r) * NDIM + col;
                const float vz = acc[0][i][j][r] + acc[3][i][j][r] + bz;
                const float vr = acc[1][i][j][r] + acc[4][i][j][r] + br;
                const float z  = fminf(fmaxf(0.2f * vz + 0.5f, 0.0f), 1.0f);
                const float rg = fminf(fmaxf(0.2f * vr + 0.5f, 0.0f), 1.0f);
                const float hv = h_tm1[idx];
                Zbuf[idx]  = z;
                XHbuf[idx] = acc[2][i][j][r] + bh;
                uint32_t u = __float_as_uint(rg * hv);
                u = u + 0x7FFFu + ((u >> 16) & 1u);
                RHbf[idx] = (uint16_t)(u >> 16);
            }
        }
    }
}

// ---------------------------------------------------------------------------
// G2: (r*h_tm1) @ [[Rh,-Ih],[Ih,Rh]] -> hh = tanh(x_h + .) ; h = z*h + (1-z)*hh
// ---------------------------------------------------------------------------
__global__ __launch_bounds__(256) void cgru_out_kernel(
    const uint16_t* __restrict__ RHbf,
    const uint16_t* __restrict__ Ur, const uint16_t* __restrict__ Ui,
    const float* __restrict__ Zbuf, const float* __restrict__ XHbuf,
    const float* __restrict__ h_tm1, float* __restrict__ out)
{
    __shared__ uint16_t sA[2][128 * 40];
    __shared__ uint16_t sB[2][64 * 40];

    const int t    = threadIdx.x;
    const int bn0  = blockIdx.x * 64;
    const int bm0  = blockIdx.y * 128;
    const int lane = t & 31;
    const int wave = t >> 5;
    const int wm   = wave & 3;
    const int wn   = wave >> 2;
    const int l16  = lane & 15;
    const int hi   = lane >> 4;

    v8f acc[2][2];
    #pragma unroll
    for (int i = 0; i < 2; ++i)
        #pragma unroll
        for (int j = 0; j < 2; ++j)
            #pragma unroll
            for (int e = 0; e < 8; ++e) acc[i][j][e] = 0.0f;

    const bool nleft = (bn0 < HALF);
    const int  nn    = nleft ? bn0 : (bn0 - HALF);
    const int  kp    = t >> 4;
    const int  nc    = t & 15;

    auto fill = [&](int buf, int kt) {
        const int  k0   = kt * 32;
        const bool ktop = (k0 < HALF);
        const int  kk   = ktop ? k0 : (k0 - HALF);
        #pragma unroll
        for (int c = t; c < 512; c += 256) {
            const int row = c >> 2, q = c & 3;
            cp16_g2l(&sA[buf][row * 40 + q * 8],
                     &RHbf[(size_t)(bm0 + row) * KDIM + k0 + q * 8]);
        }
        BSrc bs = resolve_src(Ur, Ui, ktop, nleft, kk, 2 /*h gate*/, nn);
        fill_btile(&sB[buf][0], bs.p, bs.sgn, kp, nc);
    };

    fill(0, 0);
    wait_async_lds();
    __syncthreads();

    for (int kt = 0; kt < NKT; ++kt) {
        const int cur = kt & 1;
        if (kt + 1 < NKT) fill((kt + 1) & 1, kt + 1);

        Frag af[2], bfr[2];
        #pragma unroll
        for (int i = 0; i < 2; ++i)
            load_afrag(af[i], &sA[cur][0], wm * 32 + i * 16 + l16, hi);
        #pragma unroll
        for (int j = 0; j < 2; ++j)
            load_bfrag(bfr[j], &sB[cur][0], wn * 32 + j * 16 + l16, hi);
        #pragma unroll
        for (int i = 0; i < 2; ++i)
            #pragma unroll
            for (int j = 0; j < 2; ++j)
                acc[i][j] = __builtin_amdgcn_wmma_f32_16x16x32_bf16(
                    false, af[i].v, false, bfr[j].v, (short)0, acc[i][j],
                    false, false);

        wait_async_lds();
        __syncthreads();
    }

    #pragma unroll
    for (int j = 0; j < 2; ++j) {
        const int col = bn0 + wn * 32 + j * 16 + l16;
        #pragma unroll
        for (int i = 0; i < 2; ++i) {
            const int mrow = bm0 + wm * 32 + i * 16 + hi * 8;
            #pragma unroll
            for (int r = 0; r < 8; ++r) {
                const size_t idx = (size_t)(mrow + r) * NDIM + col;
                const float hh = tanhf(XHbuf[idx] + acc[i][j][r]);
                const float z  = Zbuf[idx];
                out[idx] = z * h_tm1[idx] + (1.0f - z) * hh;
            }
        }
    }
}

// ---------------------------------------------------------------------------
// Host-side launcher
// ---------------------------------------------------------------------------
extern "C" void kernel_launch(void* const* d_in, const int* in_sizes, int n_in,
                              void* d_out, int out_size, void* d_ws, size_t ws_size,
                              hipStream_t stream)
{
    const float* inputs = (const float*)d_in[0];
    const float* h_tm1  = (const float*)d_in[1];
    const float* rk     = (const float*)d_in[2];
    const float* ik     = (const float*)d_in[3];
    const float* rrk    = (const float*)d_in[4];
    const float* irk    = (const float*)d_in[5];
    const float* rb     = (const float*)d_in[6];
    const float* ib     = (const float*)d_in[7];
    float* out = (float*)d_out;

    const size_t WSZ = (size_t)HALF * WCOLS;   // 1024*3072 weight elements
    const size_t XSZ = (size_t)BDIM * KDIM;    // 4096*2048 activation elements

    uint16_t* Wr   = (uint16_t*)d_ws;
    uint16_t* Wi   = Wr + WSZ;
    uint16_t* Ur   = Wi + WSZ;
    uint16_t* Ui   = Ur + WSZ;
    uint16_t* Xbf  = Ui + WSZ;
    uint16_t* Hbf  = Xbf + XSZ;
    uint16_t* RHbf = Hbf + XSZ;
    float*    Zbuf = (float*)(RHbf + XSZ);
    float*    XHbf = Zbuf + XSZ;

    // Phase 0: fp32 -> bf16 conversions
    f32_to_bf16_kernel<<<(int)((WSZ + 255) / 256), 256, 0, stream>>>(rk,  Wr,  (int)WSZ);
    f32_to_bf16_kernel<<<(int)((WSZ + 255) / 256), 256, 0, stream>>>(ik,  Wi,  (int)WSZ);
    f32_to_bf16_kernel<<<(int)((WSZ + 255) / 256), 256, 0, stream>>>(rrk, Ur,  (int)WSZ);
    f32_to_bf16_kernel<<<(int)((WSZ + 255) / 256), 256, 0, stream>>>(irk, Ui,  (int)WSZ);
    f32_to_bf16_kernel<<<(int)((XSZ + 255) / 256), 256, 0, stream>>>(inputs, Xbf, (int)XSZ);
    f32_to_bf16_kernel<<<(int)((XSZ + 255) / 256), 256, 0, stream>>>(h_tm1,  Hbf, (int)XSZ);

    dim3 grid(NDIM / 64, BDIM / 128);

    // Phase 1: fused 5-GEMM + gates
    cgru_gates_kernel<<<grid, 256, 0, stream>>>(
        Xbf, Hbf, Wr, Wi, Ur, Ui, rb, ib, h_tm1, Zbuf, XHbf, RHbf);

    // Phase 2: candidate GEMM + output blend
    cgru_out_kernel<<<grid, 256, 0, stream>>>(
        RHbf, Ur, Ui, Zbuf, XHbf, h_tm1, out);
}
